// AliasMHA_69200513073687
// MI455X (gfx1250) — matrix operand order repaired and checked by reading, not compiled, not run on hardware
//
#include <hip/hip_runtime.h>

// ---------------------------------------------------------------------------
// Problem constants (from the reference)
// ---------------------------------------------------------------------------
#define BB   128   // batch
#define NN   128   // sentence length
#define MM   24    // aliases
#define KC   32    // candidates per alias
#define DD   512   // model dim
#define FF   1024  // ffn dim
#define HH   8     // heads
#define DH   64    // head dim
#define NEGV (-1e9f)

typedef __bf16 bf16_t;
typedef __attribute__((ext_vector_type(16))) __bf16 v16bf;
typedef __attribute__((ext_vector_type(8)))  float  v8f;

// TDM descriptor vector types (per probe: (v4u, v8i, v4i, v4i[, v8i], i32))
typedef __attribute__((ext_vector_type(4))) unsigned tdm_v4u;
typedef __attribute__((ext_vector_type(8))) int      tdm_v8i;
typedef __attribute__((ext_vector_type(4))) int      tdm_v4i;

#ifndef AMHA_USE_TDM
#if defined(__has_builtin)
#if __has_builtin(__builtin_amdgcn_tensor_load_to_lds)
#define AMHA_USE_TDM 1
#endif
#endif
#endif
#ifndef AMHA_USE_TDM
#define AMHA_USE_TDM 0
#endif

__device__ __forceinline__ bf16_t f2bf(float f) {
  unsigned u = __builtin_bit_cast(unsigned, f);
  unsigned r = (u + 0x7fffu + ((u >> 16) & 1u)) >> 16;
  unsigned short s = (unsigned short)r;
  return __builtin_bit_cast(bf16_t, s);
}
__device__ __forceinline__ float bf2f(bf16_t b) {
  unsigned short s = __builtin_bit_cast(unsigned short, b);
  unsigned u = ((unsigned)s) << 16;
  return __builtin_bit_cast(float, u);
}

__device__ __forceinline__ void amha_wait_tensorcnt0() {
#if defined(__has_builtin) && __has_builtin(__builtin_amdgcn_s_wait_tensorcnt)
  __builtin_amdgcn_s_wait_tensorcnt(0);
#else
  asm volatile("s_wait_tensorcnt 0x0" ::: "memory");
#endif
}

#if AMHA_USE_TDM
// One TDM transfer: 128 rows x 32 bf16 tile, global row stride = strideElems,
// LDS destination rows padded 64B data + 16B pad (pad_interval=3 -> 16 DWORDs,
// pad_amount=3 -> 4 DWORDs) to match the LDS leading dim of 40 elements.
__device__ __forceinline__ void amha_tdm_load_tile(const void* gsrc,
                                                   unsigned ldsAddr,
                                                   unsigned strideElems) {
  const unsigned long long ga = (unsigned long long)(size_t)gsrc;
  const unsigned TD = 1u << 20;  // generous tensor dims (tile always in-bounds)
  tdm_v4u g0;
  g0.x = 1u;                                       // count = 1 (valid D#)
  g0.y = ldsAddr;                                  // lds_addr (bytes)
  g0.z = (unsigned)ga;                             // global_addr[31:0]
  g0.w = (unsigned)((ga >> 32) & 0x1FFFFFFu)       // global_addr[56:32]
         | 0x80000000u;                            // type = 2 ("image")
  tdm_v8i g1;
  g1[0] = (int)((1u << 16)     // data_size = 1 -> 2 bytes
              | (1u << 20)     // pad_enable
              | (3u << 22)     // pad_interval: 16 DWORDs (64B)
              | (3u << 25));   // pad_amount:   4 DWORDs (16B)
  g1[1] = (int)((TD & 0xFFFFu) << 16);                      // tensor_dim0 lo16
  g1[2] = (int)(((TD >> 16) & 0xFFFFu) | ((TD & 0xFFFFu) << 16)); // td0 hi | td1 lo
  g1[3] = (int)(((TD >> 16) & 0xFFFFu) | (32u << 16));      // td1 hi | tile_dim0=32
  g1[4] = (int)128u;                                        // tile_dim1=128, tile_dim2=0
  g1[5] = (int)strideElems;                                 // dim0_stride lo32
  g1[6] = 0;                                                // dim0_stride hi | dim1_stride lo
  g1[7] = 0;
  tdm_v4i z4 = {};
#if defined(__clang_major__) && __clang_major__ >= 23
  tdm_v8i z8 = {};
  __builtin_amdgcn_tensor_load_to_lds(g0, g1, z4, z4, z8, 0);
#else
  __builtin_amdgcn_tensor_load_to_lds(g0, g1, z4, z4, 0);
#endif
}
#endif  // AMHA_USE_TDM

// ---------------------------------------------------------------------------
// f32 -> bf16 conversion (grid-stride)
// ---------------------------------------------------------------------------
__global__ void amha_cvt_kernel(const float* __restrict__ in,
                                bf16_t* __restrict__ out, long long n) {
  long long i = (long long)blockIdx.x * blockDim.x + threadIdx.x;
  long long stride = (long long)gridDim.x * blockDim.x;
  for (; i < n; i += stride) out[i] = f2bf(in[i]);
}

// ---------------------------------------------------------------------------
// Weight transpose + cast: in[r][c] f32 -> out[c][r] bf16 (32x32 LDS tiles)
// ---------------------------------------------------------------------------
__global__ __launch_bounds__(256) void amha_transpose_kernel(
    const float* __restrict__ in, bf16_t* __restrict__ out,
    int rows, int cols) {
  __shared__ float t[32][33];
  const int c0 = blockIdx.x * 32, r0 = blockIdx.y * 32;
  const int tx = threadIdx.x & 31, ty = threadIdx.x >> 5;
  for (int i = ty; i < 32; i += 8)
    t[i][tx] = in[(size_t)(r0 + i) * cols + c0 + tx];
  __syncthreads();
  for (int i = ty; i < 32; i += 8)
    out[(size_t)(c0 + i) * rows + r0 + tx] = f2bf(t[tx][i]);
}

// ---------------------------------------------------------------------------
// Gather alias start words + slice embedding -> aw0 (f32 and bf16)
// ---------------------------------------------------------------------------
__global__ void amha_gather_kernel(const float* __restrict__ sent,
                                   const int* __restrict__ astart,
                                   const float* __restrict__ slice,
                                   float* __restrict__ awf,
                                   bf16_t* __restrict__ awb) {
  const int bm = blockIdx.x;          // b*MM + m
  const int b  = bm / MM;
  const int a  = astart[bm];
  const float* src = sent + ((size_t)b * NN + a) * DD;
  for (int d = threadIdx.x; d < DD; d += blockDim.x) {
    float v = src[d] + slice[d];
    awf[(size_t)bm * DD + d] = v;
    awb[(size_t)bm * DD + d] = f2bf(v);
  }
}

// ---------------------------------------------------------------------------
// bf16 WMMA GEMM:  C[R,Ncol] = A[R,Ka] @ Wt^T + bias  (Wt is [Ncol][Ka] bf16)
// Block tile 128x128x32, 256 threads (8 waves), each wave owns 64x32.
// A and B tiles are identical 128x32 row-major loads; staged to LDS by the
// Tensor Data Mover (one tensor_load_to_lds per tile, wave 0 issues, waits
// s_wait_tensorcnt, workgroup barrier publishes). Fallback: vector loads.
// ---------------------------------------------------------------------------
template <bool RELU>
__global__ __launch_bounds__(256) void amha_gemm_kernel(
    const bf16_t* __restrict__ A, const bf16_t* __restrict__ Wt,
    const float* __restrict__ bias,
    float* __restrict__ Cf, bf16_t* __restrict__ Cb,
    int R, int Ncol, int Ka) {
  constexpr int BM = 128, BN = 128, BK = 32, LDT = 40;
  __shared__ __align__(16) bf16_t As[2][BM][LDT];
  __shared__ __align__(16) bf16_t Bs[2][BN][LDT];

  const int tid = threadIdx.x;
  const int rowBase = blockIdx.y * BM;
  const int nBase   = blockIdx.x * BN;
  const int wid = tid >> 5, lane = tid & 31;
  const int wr = wid >> 2;          // 0..1  -> 64-row slab
  const int wc = wid & 3;           // 0..3  -> 32-col slab
  const int l15 = lane & 15, lhi = lane >> 4;

  union U32B { uint4 q[2]; v16bf v; };

  v8f acc[4][2];
#pragma unroll
  for (int i = 0; i < 4; ++i)
#pragma unroll
    for (int j = 0; j < 2; ++j) acc[i][j] = v8f{};

  const int nk = Ka / BK;
  int buf = 0;

#if AMHA_USE_TDM
  const bool issuer = (tid < 32);  // wave 0 drives the TDM
  if (issuer) {
    amha_tdm_load_tile(A  + (size_t)rowBase * Ka, (unsigned)(size_t)&As[0][0][0], Ka);
    amha_tdm_load_tile(Wt + (size_t)nBase   * Ka, (unsigned)(size_t)&Bs[0][0][0], Ka);
  }
#else
  auto loadTiles = [&](int bsel, int k0) {
    const int c0 = tid, c1 = tid + 256;
    const int r0c = c0 >> 2, k0c = (c0 & 3) * 8;
    const int r1c = c1 >> 2, k1c = (c1 & 3) * 8;
    // hoist all global loads, then store (single wait instead of four)
    uint4 a0 = *(const uint4*)(A  + (size_t)(rowBase + r0c) * Ka + k0 + k0c);
    uint4 a1 = *(const uint4*)(A  + (size_t)(rowBase + r1c) * Ka + k0 + k1c);
    uint4 b0 = *(const uint4*)(Wt + (size_t)(nBase   + r0c) * Ka + k0 + k0c);
    uint4 b1 = *(const uint4*)(Wt + (size_t)(nBase   + r1c) * Ka + k0 + k1c);
    *(uint4*)&As[bsel][r0c][k0c] = a0;
    *(uint4*)&As[bsel][r1c][k1c] = a1;
    *(uint4*)&Bs[bsel][r0c][k0c] = b0;
    *(uint4*)&Bs[bsel][r1c][k1c] = b1;
  };
  loadTiles(0, 0);
#endif

  for (int kt = 0; kt < nk; ++kt) {
#if AMHA_USE_TDM
    if (issuer) amha_wait_tensorcnt0();  // my TDM transfers into `buf` landed
    __syncthreads();                      // publish to all 8 waves
    if (kt + 1 < nk && issuer) {
      const int k1 = (kt + 1) * BK;
      amha_tdm_load_tile(A  + (size_t)rowBase * Ka + k1,
                         (unsigned)(size_t)&As[buf ^ 1][0][0], Ka);
      amha_tdm_load_tile(Wt + (size_t)nBase   * Ka + k1,
                         (unsigned)(size_t)&Bs[buf ^ 1][0][0], Ka);
    }
#else
    __syncthreads();
    if (kt + 1 < nk) loadTiles(buf ^ 1, (kt + 1) * BK);
    if (kt + 2 < nk) {  // L2 prefetch (global_prefetch_b8)
      __builtin_prefetch(A  + (size_t)(rowBase + (tid & 127)) * Ka + (kt + 2) * BK, 0, 0);
      __builtin_prefetch(Wt + (size_t)(nBase   + (tid & 127)) * Ka + (kt + 2) * BK, 0, 0);
    }
#endif
    // A fragments: lanes 0-15 row m: elems 0..7 = K0..7, 8..15 = K16..23;
    //              lanes 16-31 same row: K8..15 / K24..31.
    v16bf afrag[4];
#pragma unroll
    for (int i = 0; i < 4; ++i) {
      const bf16_t* p = &As[buf][wr * 64 + i * 16 + l15][lhi * 8];
      U32B u;
      u.q[0] = *(const uint4*)p;
      u.q[1] = *(const uint4*)(p + 16);
      afrag[i] = u.v;
    }
    // B fragments: lane holds column n = l15; elems 0..7 = K lhi*8..+7, 8..15 = K 16+lhi*8..+7.
    v16bf bfrag[2];
#pragma unroll
    for (int j = 0; j < 2; ++j) {
      const bf16_t* p = &Bs[buf][wc * 32 + j * 16 + l15][lhi * 8];
      U32B u;
      u.q[0] = *(const uint4*)p;
      u.q[1] = *(const uint4*)(p + 16);
      bfrag[j] = u.v;
    }
#pragma unroll
    for (int i = 0; i < 4; ++i)
#pragma unroll
      for (int j = 0; j < 2; ++j)
        acc[i][j] = __builtin_amdgcn_wmma_f32_16x16x32_bf16(
            false, afrag[i], false, bfrag[j], (short)0, acc[i][j], false, false);
    buf ^= 1;
  }

  // Epilogue: C/D layout -> row = base + lhi*8 + r, col = base + l15
  const int colB = nBase + wc * 32;
#pragma unroll
  for (int j = 0; j < 2; ++j) {
    const int col = colB + j * 16 + l15;
    const float bv = bias ? bias[col] : 0.f;
#pragma unroll
    for (int i = 0; i < 4; ++i) {
      const int row0 = rowBase + wr * 64 + i * 16 + lhi * 8;
#pragma unroll
      for (int r = 0; r < 8; ++r) {
        float x = acc[i][j][r] + bv;
        if (RELU) x = fmaxf(x, 0.f);
        const size_t idx = (size_t)(row0 + r) * Ncol + col;
        if (Cf) Cf[idx] = x;
        if (Cb) Cb[idx] = f2bf(x);
      }
    }
  }
}

// ---------------------------------------------------------------------------
// Attention block 1 core: per-batch WG, loops heads. Q:[24,64] K,V:[128,64].
// ---------------------------------------------------------------------------
__global__ __launch_bounds__(256) void amha_attn1_kernel(
    const float* __restrict__ Q, const float* __restrict__ Kt,
    const float* __restrict__ Vt, const unsigned char* __restrict__ smask,
    bf16_t* __restrict__ ctxb, float* __restrict__ aww) {
  const int b = blockIdx.x, tid = threadIdx.x;
  __shared__ float sQ[MM * DH];
  __shared__ float sS[MM * NN];
  __shared__ float sW[MM * NN];
  for (int i = tid; i < MM * NN; i += 256) sW[i] = 0.f;
  const float scale = 0.125f;  // 1/sqrt(64)
  for (int h = 0; h < HH; ++h) {
    __syncthreads();
    for (int i = tid; i < MM * DH; i += 256) {
      const int m = i >> 6, j = i & 63;
      sQ[i] = Q[((size_t)(b * MM + m)) * DD + h * DH + j];
    }
    __syncthreads();
    for (int i = tid; i < MM * NN; i += 256) {
      const int m = i >> 7, k = i & 127;
      const float* kp = Kt + ((size_t)(b * NN + k)) * DD + h * DH;
      const float* qp = sQ + m * DH;
      float s = 0.f;
#pragma unroll 8
      for (int j = 0; j < DH; ++j) s += qp[j] * kp[j];
      s *= scale;
      if (smask[b * NN + k]) s = NEGV;
      sS[i] = s;
    }
    __syncthreads();
    if (tid < MM) {  // softmax per query row
      float* row = sS + tid * NN;
      float mx = row[0];
      for (int k = 1; k < NN; ++k) mx = fmaxf(mx, row[k]);
      float sum = 0.f;
      for (int k = 0; k < NN; ++k) { float e = __expf(row[k] - mx); row[k] = e; sum += e; }
      const float inv = 1.f / sum;
      for (int k = 0; k < NN; ++k) row[k] *= inv;
    }
    __syncthreads();
    for (int i = tid; i < MM * NN; i += 256) sW[i] += sS[i] * (1.f / HH);
    for (int i = tid; i < MM * DH; i += 256) {
      const int m = i >> 6, j = i & 63;
      const float* w = sS + m * NN;
      float acc2 = 0.f;
      for (int k = 0; k < NN; ++k)
        acc2 += w[k] * Vt[((size_t)(b * NN + k)) * DD + h * DH + j];
      ctxb[((size_t)(b * MM + m)) * DD + h * DH + j] = f2bf(acc2);
    }
  }
  __syncthreads();
  for (int i = tid; i < MM * NN; i += 256)
    aww[(size_t)b * MM * NN + i] = sW[i];
}

// ---------------------------------------------------------------------------
// Attention block 2 core: block-diagonal; WG per (b,m); keys = 32 candidates.
// ---------------------------------------------------------------------------
__global__ __launch_bounds__(256) void amha_attn2_kernel(
    const float* __restrict__ Q, const bf16_t* __restrict__ Kb,
    const bf16_t* __restrict__ Vb, const unsigned char* __restrict__ emask,
    bf16_t* __restrict__ ctxb) {
  const int bm = blockIdx.x;
  const int b = bm / MM, m = bm % MM;
  const int tid = threadIdx.x;
  __shared__ float sQ[DD];
  __shared__ float sS[HH][KC];
  const size_t kbase = (size_t)b * (MM * KC) + m * KC;
  for (int i = tid; i < DD; i += 256) sQ[i] = Q[(size_t)bm * DD + i];
  __syncthreads();
  {
    const int h = tid >> 5, k = tid & 31;
    const bf16_t* kp = Kb + (kbase + k) * DD + h * DH;
    const float* qp = sQ + h * DH;
    float s = 0.f;
#pragma unroll 8
    for (int j = 0; j < DH; ++j) s += qp[j] * bf2f(kp[j]);
    s *= 0.125f;
    if (emask[kbase + k]) s = NEGV;
    sS[h][k] = s;
  }
  __syncthreads();
  if (tid < HH) {
    float* row = sS[tid];
    float mx = row[0];
    for (int k = 1; k < KC; ++k) mx = fmaxf(mx, row[k]);
    float sum = 0.f;
    for (int k = 0; k < KC; ++k) { float e = __expf(row[k] - mx); row[k] = e; sum += e; }
    const float inv = 1.f / sum;
    for (int k = 0; k < KC; ++k) row[k] *= inv;
  }
  __syncthreads();
  for (int d = tid; d < DD; d += 256) {
    const int h = d >> 6;
    const float* w = sS[h];
    float acc2 = 0.f;
#pragma unroll 8
    for (int k = 0; k < KC; ++k) acc2 += w[k] * bf2f(Vb[(kbase + k) * DD + d]);
    ctxb[(size_t)bm * DD + d] = f2bf(acc2);
  }
}

// ---------------------------------------------------------------------------
// LayerNorm(x + y) * g + b  (+ optional addvec), width 512, WG per row.
// ---------------------------------------------------------------------------
__global__ __launch_bounds__(256) void amha_ln_kernel(
    const float* __restrict__ x, const float* __restrict__ y,
    const float* __restrict__ g, const float* __restrict__ be,
    const float* __restrict__ addv,
    float* __restrict__ outf, bf16_t* __restrict__ outb) {
  const int row = blockIdx.x, tid = threadIdx.x;
  __shared__ float red[256];
  const size_t base = (size_t)row * DD;
  const float z0 = x[base + tid] + y[base + tid];
  const float z1 = x[base + 256 + tid] + y[base + 256 + tid];
  red[tid] = z0 + z1;
  __syncthreads();
  for (int s = 128; s > 0; s >>= 1) {
    if (tid < s) red[tid] += red[tid + s];
    __syncthreads();
  }
  const float mu = red[0] * (1.f / DD);
  __syncthreads();
  const float d0 = z0 - mu, d1 = z1 - mu;
  red[tid] = d0 * d0 + d1 * d1;
  __syncthreads();
  for (int s = 128; s > 0; s >>= 1) {
    if (tid < s) red[tid] += red[tid + s];
    __syncthreads();
  }
  const float rstd = rsqrtf(red[0] * (1.f / DD) + 1e-5f);
  float o0 = d0 * rstd * g[tid] + be[tid];
  float o1 = d1 * rstd * g[256 + tid] + be[256 + tid];
  if (addv) { o0 += addv[tid]; o1 += addv[256 + tid]; }
  if (outf) { outf[base + tid] = o0; outf[base + 256 + tid] = o1; }
  if (outb) { outb[base + tid] = f2bf(o0); outb[base + 256 + tid] = f2bf(o1); }
}

// ---------------------------------------------------------------------------
// Host orchestration
// ---------------------------------------------------------------------------
struct AmhaBlk {
  const float *Wq, *Wk, *Wv, *bq, *bk, *bv, *Wo, *bo;
  const float *g1, *be1, *g2, *be2, *W1, *bb1, *W2, *bb2;
};

extern "C" void kernel_launch(void* const* d_in, const int* in_sizes, int n_in,
                              void* d_out, int out_size, void* d_ws, size_t ws_size,
                              hipStream_t stream) {
  (void)in_sizes; (void)n_in; (void)out_size; (void)ws_size;

  const float* sent          = (const float*)d_in[0];
  const unsigned char* smask = (const unsigned char*)d_in[1];
  const float* ent           = (const float*)d_in[2];
  const unsigned char* emask = (const unsigned char*)d_in[3];
  const int* astart          = (const int*)d_in[4];
  const float* sl_alias      = (const float*)d_in[5];
  const float* sl_ent        = (const float*)d_in[6];

  auto getBlk = [&](int base) {
    AmhaBlk p;
    p.Wq = (const float*)d_in[base + 0];  p.Wk  = (const float*)d_in[base + 1];
    p.Wv = (const float*)d_in[base + 2];  p.bq  = (const float*)d_in[base + 3];
    p.bk = (const float*)d_in[base + 4];  p.bv  = (const float*)d_in[base + 5];
    p.Wo = (const float*)d_in[base + 6];  p.bo  = (const float*)d_in[base + 7];
    p.g1 = (const float*)d_in[base + 8];  p.be1 = (const float*)d_in[base + 9];
    p.g2 = (const float*)d_in[base + 10]; p.be2 = (const float*)d_in[base + 11];
    p.W1 = (const float*)d_in[base + 12]; p.bb1 = (const float*)d_in[base + 13];
    p.W2 = (const float*)d_in[base + 14]; p.bb2 = (const float*)d_in[base + 15];
    return p;
  };
  const AmhaBlk sa = getBlk(7);
  const AmhaBlk ea = getBlk(23);

  const long long RA = (long long)BB * MM;        // 3072 alias rows
  const long long RS = (long long)BB * NN;        // 16384 sentence rows
  const long long RE = (long long)BB * MM * KC;   // 98304 entity rows

  size_t off = 0;
  auto alloc = [&](size_t bytes) {
    void* p = (char*)d_ws + off;
    off += (bytes + 255) & ~(size_t)255;
    return p;
  };
  bf16_t* sent_bf = (bf16_t*)alloc(RS * DD * 2);
  bf16_t* ent_bf  = (bf16_t*)alloc(RE * DD * 2);
  bf16_t* wb[12];
  const size_t wsz[12] = {DD*DD, DD*DD, DD*DD, DD*DD, (size_t)DD*FF, (size_t)FF*DD,
                          DD*DD, DD*DD, DD*DD, DD*DD, (size_t)DD*FF, (size_t)FF*DD};
  for (int i = 0; i < 12; ++i) wb[i] = (bf16_t*)alloc(wsz[i] * 2);
  float*  aw0f  = (float*)alloc(RA * DD * 4);
  bf16_t* aw0b  = (bf16_t*)alloc(RA * DD * 2);
  float*  Q1    = (float*)alloc(RA * DD * 4);
  float*  K1    = (float*)alloc(RS * DD * 4);
  float*  V1    = (float*)alloc(RS * DD * 4);
  bf16_t* ctx1b = (bf16_t*)alloc(RA * DD * 2);
  float*  o1    = (float*)alloc(RA * DD * 4);
  float*  res1f = (float*)alloc(RA * DD * 4);
  bf16_t* res1b = (bf16_t*)alloc(RA * DD * 2);
  bf16_t* h1b   = (bf16_t*)alloc(RA * FF * 2);
  float*  ff1   = (float*)alloc(RA * DD * 4);
  float*  q2f   = (float*)alloc(RA * DD * 4);
  bf16_t* q2b   = (bf16_t*)alloc(RA * DD * 2);
  float*  Q2    = (float*)alloc(RA * DD * 4);
  bf16_t* K2b   = (bf16_t*)alloc(RE * DD * 2);
  bf16_t* V2b   = (bf16_t*)alloc(RE * DD * 2);
  bf16_t* ctx2b = (bf16_t*)alloc(RA * DD * 2);
  float*  o2    = (float*)alloc(RA * DD * 4);
  float*  res2f = (float*)alloc(RA * DD * 4);
  bf16_t* res2b = (bf16_t*)alloc(RA * DD * 2);
  bf16_t* h2b   = (bf16_t*)alloc(RA * FF * 2);
  float*  ff2   = (float*)alloc(RA * DD * 4);

  float* out_ctx = (float*)d_out;
  float* out_aww = (float*)d_out + (size_t)RA * DD;

  auto cvt = [&](const float* in, bf16_t* out, long long n) {
    long long b = (n + 255) / 256; if (b > 4096) b = 4096;
    amha_cvt_kernel<<<dim3((unsigned)b), dim3(256), 0, stream>>>(in, out, n);
  };
  auto gemm = [&](const bf16_t* A, const bf16_t* Wt, const float* bias,
                  float* Cf, bf16_t* Cb, int R, int Ncol, int Ka, bool relu) {
    dim3 g(Ncol / 128, R / 128), blk(256);
    if (relu) amha_gemm_kernel<true ><<<g, blk, 0, stream>>>(A, Wt, bias, Cf, Cb, R, Ncol, Ka);
    else      amha_gemm_kernel<false><<<g, blk, 0, stream>>>(A, Wt, bias, Cf, Cb, R, Ncol, Ka);
  };
  auto ln = [&](const float* x, const float* y, const float* g, const float* be,
                const float* addv, float* outf, bf16_t* outb, int rows) {
    amha_ln_kernel<<<dim3(rows), dim3(256), 0, stream>>>(x, y, g, be, addv, outf, outb);
  };

  // ---- activation casts + weight transpose (W[k][n] f32 -> Wt[n][k] bf16) --
  cvt(sent, sent_bf, RS * DD);
  cvt(ent,  ent_bf,  RE * DD);
  struct { const float* src; int rows, cols; } wt[12] = {
      {sa.Wq, DD, DD}, {sa.Wk, DD, DD}, {sa.Wv, DD, DD}, {sa.Wo, DD, DD},
      {sa.W1, DD, FF}, {sa.W2, FF, DD},
      {ea.Wq, DD, DD}, {ea.Wk, DD, DD}, {ea.Wv, DD, DD}, {ea.Wo, DD, DD},
      {ea.W1, DD, FF}, {ea.W2, FF, DD}};
  for (int i = 0; i < 12; ++i)
    amha_transpose_kernel<<<dim3(wt[i].cols / 32, wt[i].rows / 32), dim3(256), 0, stream>>>(
        wt[i].src, wb[i], wt[i].rows, wt[i].cols);

  // ---- gather + slice_emb_alias ----------------------------------------
  amha_gather_kernel<<<dim3((unsigned)RA), dim3(256), 0, stream>>>(
      sent, astart, sl_alias, aw0f, aw0b);

  // ---- block 1 (alias -> sentence) --------------------------------------
  gemm(aw0b,   wb[0], sa.bq, Q1, nullptr, (int)RA, DD, DD, false);
  gemm(sent_bf,wb[1], sa.bk, K1, nullptr, (int)RS, DD, DD, false);
  gemm(sent_bf,wb[2], sa.bv, V1, nullptr, (int)RS, DD, DD, false);
  amha_attn1_kernel<<<dim3(BB), dim3(256), 0, stream>>>(Q1, K1, V1, smask, ctx1b, out_aww);
  gemm(ctx1b,  wb[3], sa.bo, o1, nullptr, (int)RA, DD, DD, false);
  ln(aw0f, o1, sa.g1, sa.be1, nullptr, res1f, res1b, (int)RA);
  gemm(res1b,  wb[4], sa.bb1, nullptr, h1b, (int)RA, FF, DD, true);   // relu
  gemm(h1b,    wb[5], sa.bb2, ff1, nullptr, (int)RA, DD, FF, false);
  ln(res1f, ff1, sa.g2, sa.be2, sl_ent, q2f, q2b, (int)RA);           // + slice_emb_ent

  // ---- block 2 (alias -> entities, block-diagonal) -----------------------
  gemm(q2b,    wb[6], ea.bq, Q2, nullptr, (int)RA, DD, DD, false);
  gemm(ent_bf, wb[7], ea.bk, nullptr, K2b, (int)RE, DD, DD, false);
  gemm(ent_bf, wb[8], ea.bv, nullptr, V2b, (int)RE, DD, DD, false);
  amha_attn2_kernel<<<dim3((unsigned)RA), dim3(256), 0, stream>>>(Q2, K2b, V2b, emask, ctx2b);
  gemm(ctx2b,  wb[9], ea.bo, o2, nullptr, (int)RA, DD, DD, false);
  ln(q2f, o2, ea.g1, ea.be1, nullptr, res2f, res2b, (int)RA);
  gemm(res2b,  wb[10], ea.bb1, nullptr, h2b, (int)RA, FF, DD, true);  // relu
  gemm(h2b,    wb[11], ea.bb2, ff2, nullptr, (int)RA, DD, FF, false);
  ln(res2f, ff2, ea.g2, ea.be2, nullptr, out_ctx, nullptr, (int)RA);  // final output
}
